// Decoder_78434692760059
// MI455X (gfx1250) — compile-verified, matrix operand order checked by reading, not run on hardware
//
#include <hip/hip_runtime.h>
#include <hip/hip_bf16.h>

// ---------------- problem constants ----------------
#define B   256
#define TK  400
#define H   256
#define E   128
#define V   50000
#define OOV 50
#define N   512          // 2H
#define G4H 1024         // 4H
#define VO  (V + OOV)    // 50050

// ---------------- d_out offsets (floats) ----------------
#define O_FINAL 0
#define O_HT    (B * VO)
#define O_CC    (O_HT + B * H)
#define O_CT    (O_CC + B * H)
#define O_ATTN  (O_CT + B * N)
#define O_PGEN  (O_ATTN + B * TK)
#define O_COV   (O_PGEN + B)

// ---------------- ws float-region offsets (floats) ----------------
#define W_X      0                       // [B,E]
#define W_GATES  (W_X + B * E)           // [B,4H]
#define W_DECF   (W_GATES + B * G4H)     // [B,N]
#define W_SCORES (W_DECF + B * N)        // [B,TK]
#define W_RMAX   (W_SCORES + B * TK)     // [B]
#define W_RSUM   (W_RMAX + B)            // [B]
#define W_LOGITS (W_RSUM + B)            // [B,V]
#define WS_FLOATS_END (W_LOGITS + (size_t)B * V)
// ---------------- ws half-region offsets (halves, after float region) ---
#define Hx_WH16   0                      // [N,N]      f16 W_h
#define Hx_WO216  (Hx_WH16 + N * N)      // [V,H]      f16 W_o2
#define Hx_OUT16  (Hx_WO216 + (size_t)V * H)  // [B,H] f16 out

typedef _Float16 v16h __attribute__((ext_vector_type(16)));
typedef _Float16 v8h  __attribute__((ext_vector_type(8)));
typedef _Float16 v4h  __attribute__((ext_vector_type(4)));
typedef float    v8f  __attribute__((ext_vector_type(8)));

__device__ __forceinline__ float sigmf(float x) { return 1.0f / (1.0f + __expf(-x)); }

// f32 source fragment: elements 0..7 <- p[0..7], 8..15 <- p[16..23]
__device__ __forceinline__ v16h ld_frag_f32(const float* __restrict__ p) {
    v16h r;
#pragma unroll
    for (int i = 0; i < 8; ++i) {
        r[i]     = (_Float16)p[i];
        r[i + 8] = (_Float16)p[16 + i];
    }
    return r;
}

// f16 source fragment: two aligned 16B loads -> global_load_b128
__device__ __forceinline__ v16h ld_frag_f16(const _Float16* __restrict__ p) {
    v8h lo = *(const v8h*)p;
    v8h hi = *(const v8h*)(p + 16);
    v16h r;
#pragma unroll
    for (int i = 0; i < 8; ++i) { r[i] = lo[i]; r[i + 8] = hi[i]; }
    return r;
}

// ============ K0: f32 -> f16 weight conversion (vectorized x4) ============
__global__ void k_cvt4(const float* __restrict__ src, _Float16* __restrict__ dst, int n4) {
    int idx = blockIdx.x * blockDim.x + threadIdx.x;
    if (idx >= n4) return;
    float4 v = ((const float4*)src)[idx];
    v4h h;
    h[0] = (_Float16)v.x; h[1] = (_Float16)v.y;
    h[2] = (_Float16)v.z; h[3] = (_Float16)v.w;
    ((v4h*)dst)[idx] = h;
}

// ============ K1: x = [c_t_1 | emb[y]] @ Wx.T + bx ============
__global__ void k_xproj(const int* __restrict__ y, const float* __restrict__ c_t_1,
                        const float* __restrict__ emb, const float* __restrict__ Wx,
                        const float* __restrict__ bx, float* __restrict__ xo) {
    int idx = blockIdx.x * blockDim.x + threadIdx.x;
    if (idx >= B * E) return;
    int b = idx / E, e = idx % E;
    const float* wr = Wx + (size_t)e * (N + E);
    float acc = bx[e];
    const float* c = c_t_1 + (size_t)b * N;
    for (int j = 0; j < N; ++j) acc += c[j] * wr[j];
    const float* em = emb + (size_t)y[b] * E;
    for (int j = 0; j < E; ++j) acc += em[j] * wr[N + j];
    xo[idx] = acc;
}

// ============ K2a: gates ============
__global__ void k_gates(const float* __restrict__ x, const float* __restrict__ h0,
                        const float* __restrict__ Wih, const float* __restrict__ Whh,
                        const float* __restrict__ bih, const float* __restrict__ bhh,
                        float* __restrict__ gates) {
    int idx = blockIdx.x * blockDim.x + threadIdx.x;
    if (idx >= B * G4H) return;
    int b = idx / G4H, g = idx % G4H;
    float acc = bih[g] + bhh[g];
    const float* wi = Wih + (size_t)g * E;
    const float* xr = x + (size_t)b * E;
    for (int j = 0; j < E; ++j) acc += xr[j] * wi[j];
    const float* wh = Whh + (size_t)g * H;
    const float* hr = h0 + (size_t)b * H;
    for (int j = 0; j < H; ++j) acc += hr[j] * wh[j];
    gates[idx] = acc;
}

// ============ K2b: LSTM pointwise ============
__global__ void k_lstm(const float* __restrict__ gates, const float* __restrict__ c0,
                       float* __restrict__ h_t, float* __restrict__ c_cell) {
    int idx = blockIdx.x * blockDim.x + threadIdx.x;
    if (idx >= B * H) return;
    int b = idx / H, h = idx % H;
    const float* g = gates + (size_t)b * G4H;
    float ig = g[h], fg = g[H + h], gg = g[2 * H + h], og = g[3 * H + h];
    float c = sigmf(fg) * c0[idx] + sigmf(ig) * tanhf(gg);
    h_t[idx]    = sigmf(og) * tanhf(c);
    c_cell[idx] = c;
}

// ============ K2c: dec_fea = s_t_hat @ W_dp.T + b_dp ============
__global__ void k_decfea(const float* __restrict__ h_t, const float* __restrict__ c_cell,
                         const float* __restrict__ Wdp, const float* __restrict__ bdp,
                         float* __restrict__ decf) {
    int idx = blockIdx.x * blockDim.x + threadIdx.x;
    if (idx >= B * N) return;
    int b = idx / N, n = idx % N;
    const float* wr = Wdp + (size_t)n * N;
    float acc = bdp[n];
    const float* hr = h_t + (size_t)b * H;
    const float* cr = c_cell + (size_t)b * H;
    for (int j = 0; j < H; ++j) acc += hr[j] * wr[j];
    for (int j = 0; j < H; ++j) acc += cr[j] * wr[H + j];
    decf[idx] = acc;
}

// ============ K3: attention scores (WMMA, A tile cached in VGPRs) ============
// one wave per 16-row (b,t) tile; 6400 waves, 4 waves/block -> 1600 blocks
// __launch_bounds__(128, 1): relax VGPR budget so the 16 A fragments
// (128 VGPRs) stay register-resident without spilling.
__global__ void __launch_bounds__(128, 1)
k_scores(const float* __restrict__ enc, const _Float16* __restrict__ wh16,
         const float* __restrict__ decf, const float* __restrict__ W_c,
         const float* __restrict__ v_w, const float* __restrict__ cover,
         float* __restrict__ scores) {
    int wave = (blockIdx.x * blockDim.x + threadIdx.x) >> 5;
    int lane = threadIdx.x & 31;
    int b  = wave / (TK / 16);
    int t0 = (wave % (TK / 16)) * 16;
    int half = lane >> 4, q = lane & 15, koff = half * 8;

    // load + convert the whole 16x512 A tile into registers (16 fragments)
    const float* arow = enc + ((size_t)b * TK + t0 + q) * N + koff;
    v16h afr[16];
#pragma unroll
    for (int ks = 0; ks < 16; ++ks) afr[ks] = ld_frag_f32(arow + ks * 32);

    float cov[8];
#pragma unroll
    for (int r = 0; r < 8; ++r) cov[r] = cover[b * TK + t0 + half * 8 + r];

    float acc[8];
#pragma unroll
    for (int r = 0; r < 8; ++r) acc[r] = 0.0f;

    for (int nt = 0; nt < N / 16; ++nt) {
        int n = nt * 16 + q;
        const _Float16* brow = wh16 + (size_t)n * N + koff;
        if (nt + 1 < N / 16)
            __builtin_prefetch(wh16 + (size_t)(n + 16) * N + koff, 0, 1);
        v8f c;
#pragma unroll
        for (int i = 0; i < 8; ++i) c[i] = 0.0f;
#pragma unroll
        for (int ks = 0; ks < 16; ++ks) {
            v16h bb = ld_frag_f16(brow + ks * 32);
            c = __builtin_amdgcn_wmma_f32_16x16x32_f16(false, afr[ks], false, bb,
                                                       (short)0, c, false, false);
        }
        float dn = decf[b * N + n];
        float wc = W_c[n];
        float vv = v_w[n];
#pragma unroll
        for (int r = 0; r < 8; ++r) {
            float t = tanhf(c[r] + dn + cov[r] * wc);
            acc[r] += t * vv;
        }
    }
#pragma unroll
    for (int r = 0; r < 8; ++r) {
        float s = acc[r];
        s += __shfl_xor(s, 1);
        s += __shfl_xor(s, 2);
        s += __shfl_xor(s, 4);
        s += __shfl_xor(s, 8);
        acc[r] = s;
    }
    if (q == 0) {
#pragma unroll
        for (int r = 0; r < 8; ++r) scores[b * TK + t0 + half * 8 + r] = acc[r];
    }
}

// ============ K4: masked softmax over TK + coverage update ============
__global__ void k_attn(const float* __restrict__ scores, const float* __restrict__ mask,
                       const float* __restrict__ cover, float* __restrict__ attn,
                       float* __restrict__ cov_out) {
    int b = blockIdx.x, tid = threadIdx.x;
    __shared__ float red[256];
    float m = -1e30f;
    for (int t = tid; t < TK; t += 256) m = fmaxf(m, scores[b * TK + t]);
    red[tid] = m; __syncthreads();
    for (int s = 128; s > 0; s >>= 1) { if (tid < s) red[tid] = fmaxf(red[tid], red[tid + s]); __syncthreads(); }
    m = red[0]; __syncthreads();
    float sum = 0.0f;
    for (int t = tid; t < TK; t += 256)
        sum += __expf(scores[b * TK + t] - m) * mask[b * TK + t];
    red[tid] = sum; __syncthreads();
    for (int s = 128; s > 0; s >>= 1) { if (tid < s) red[tid] += red[tid + s]; __syncthreads(); }
    float inv = 1.0f / red[0];
    for (int t = tid; t < TK; t += 256) {
        float a = __expf(scores[b * TK + t] - m) * mask[b * TK + t] * inv;
        attn[b * TK + t]    = a;
        cov_out[b * TK + t] = cover[b * TK + t] + a;
    }
}

// ============ K5: c_t = attn @ encoder_outputs ============
__global__ void k_ctx(const float* __restrict__ attn, const float* __restrict__ enc,
                      float* __restrict__ c_t) {
    int idx = blockIdx.x * blockDim.x + threadIdx.x;
    if (idx >= B * N) return;
    int b = idx / N, n = idx % N;
    float acc = 0.0f;
    const float* er = enc + (size_t)b * TK * N + n;
    const float* ar = attn + b * TK;
    for (int t = 0; t < TK; ++t) acc += ar[t] * er[(size_t)t * N];
    c_t[idx] = acc;
}

// ============ K6: p_gen ============
__global__ void k_pgen(const float* __restrict__ c_t, const float* __restrict__ h_t,
                       const float* __restrict__ c_cell, const float* __restrict__ x,
                       const float* __restrict__ Wpg, const float* __restrict__ bpg,
                       float* __restrict__ pgen) {
    int b = blockIdx.x, tid = threadIdx.x;
    __shared__ float red[256];
    float acc = 0.0f;
    for (int j = tid; j < N; j += 256) acc += c_t[b * N + j] * Wpg[j];
    for (int j = tid; j < H; j += 256) acc += h_t[b * H + j] * Wpg[N + j];
    for (int j = tid; j < H; j += 256) acc += c_cell[b * H + j] * Wpg[N + H + j];
    for (int j = tid; j < E; j += 256) acc += x[b * E + j] * Wpg[G4H + j];
    red[tid] = acc; __syncthreads();
    for (int s = 128; s > 0; s >>= 1) { if (tid < s) red[tid] += red[tid + s]; __syncthreads(); }
    if (tid == 0) pgen[b] = sigmf(red[0] + bpg[0]);
}

// ============ K7: out = [h_t | c_t] @ W_o1.T + b_o1  (writes f16) ============
__global__ void k_outo1(const float* __restrict__ h_t, const float* __restrict__ c_t,
                        const float* __restrict__ Wo1, const float* __restrict__ bo1,
                        _Float16* __restrict__ out16) {
    int idx = blockIdx.x * blockDim.x + threadIdx.x;
    if (idx >= B * H) return;
    int b = idx / H, o = idx % H;
    const float* wr = Wo1 + (size_t)o * (3 * H);
    float acc = bo1[o];
    const float* hr = h_t + (size_t)b * H;
    const float* cr = c_t + (size_t)b * N;
    for (int j = 0; j < H; ++j) acc += hr[j] * wr[j];
    for (int j = 0; j < N; ++j) acc += cr[j] * wr[H + j];
    out16[idx] = (_Float16)acc;
}

// ============ K8: logits = out @ W_o2.T + b_o2 (WMMA, all-f16 operands) ============
// one wave per 16x16 output tile; 50000 waves -> 6250 blocks
__global__ void k_logits(const _Float16* __restrict__ out16, const _Float16* __restrict__ wo216,
                         const float* __restrict__ bo2, float* __restrict__ logits) {
    int wave = (blockIdx.x * blockDim.x + threadIdx.x) >> 5;
    int lane = threadIdx.x & 31;
    int mt = wave & 15;
    int nt = wave >> 4;
    int half = lane >> 4, q = lane & 15, koff = half * 8;

    const _Float16* arow = out16 + (size_t)(mt * 16 + q) * H + koff;
    int n = nt * 16 + q;
    const _Float16* brow = wo216 + (size_t)n * H + koff;

    v8f c;
#pragma unroll
    for (int i = 0; i < 8; ++i) c[i] = 0.0f;
#pragma unroll
    for (int ks = 0; ks < H / 32; ++ks) {
        v16h a  = ld_frag_f16(arow + ks * 32);
        v16h bb = ld_frag_f16(brow + ks * 32);
        c = __builtin_amdgcn_wmma_f32_16x16x32_f16(false, a, false, bb,
                                                   (short)0, c, false, false);
    }
    float bias = bo2[n];
#pragma unroll
    for (int r = 0; r < 8; ++r) {
        int m = mt * 16 + half * 8 + r;
        logits[(size_t)m * V + n] = c[r] + bias;
    }
}

// ============ K9: per-row max & sum(exp) over V ============
__global__ void k_vstats(const float* __restrict__ logits, float* __restrict__ rmax,
                         float* __restrict__ rsum) {
    int b = blockIdx.x, tid = threadIdx.x;
    __shared__ float red[256];
    const float* lr = logits + (size_t)b * V;
    float m = -1e30f;
    for (int v = tid; v < V; v += 256) m = fmaxf(m, lr[v]);
    red[tid] = m; __syncthreads();
    for (int s = 128; s > 0; s >>= 1) { if (tid < s) red[tid] = fmaxf(red[tid], red[tid + s]); __syncthreads(); }
    m = red[0]; __syncthreads();
    float sum = 0.0f;
    for (int v = tid; v < V; v += 256) sum += __expf(lr[v] - m);
    red[tid] = sum; __syncthreads();
    for (int s = 128; s > 0; s >>= 1) { if (tid < s) red[tid] += red[tid + s]; __syncthreads(); }
    if (tid == 0) { rmax[b] = m; rsum[b] = red[0]; }
}

// ============ K11: final_dist base ============
__global__ void k_final(const float* __restrict__ logits, const float* __restrict__ rmax,
                        const float* __restrict__ rsum, const float* __restrict__ pgen,
                        const float* __restrict__ xz, float* __restrict__ fin) {
    long long idx = (long long)blockIdx.x * blockDim.x + threadIdx.x;
    if (idx >= (long long)B * VO) return;
    int b = (int)(idx / VO), v = (int)(idx % VO);
    float r;
    if (v < V) r = pgen[b] * __expf(logits[(size_t)b * V + v] - rmax[b]) / rsum[b];
    else       r = xz[b * OOV + (v - V)];
    fin[idx] = r;
}

// ============ K12: scatter-add ============
__global__ void k_scatter(const int* __restrict__ ev, const float* __restrict__ attn,
                          const float* __restrict__ pgen, float* __restrict__ fin) {
    int idx = blockIdx.x * blockDim.x + threadIdx.x;
    if (idx >= B * TK) return;
    int b = idx / TK;
    float val = (1.0f - pgen[b]) * attn[idx];
    __hip_atomic_fetch_add(&fin[(size_t)b * VO + ev[idx]], val,
                           __ATOMIC_RELAXED, __HIP_MEMORY_SCOPE_AGENT);
}

extern "C" void kernel_launch(void* const* d_in, const int* in_sizes, int n_in,
                              void* d_out, int out_size, void* d_ws, size_t ws_size,
                              hipStream_t stream) {
    const int*   y_t_1  = (const int*)  d_in[0];
    const float* h0     = (const float*)d_in[1];
    const float* c0     = (const float*)d_in[2];
    const float* enc    = (const float*)d_in[3];
    const float* mask   = (const float*)d_in[4];
    const float* c_t_1  = (const float*)d_in[5];
    const float* xz     = (const float*)d_in[6];
    const int*   ev     = (const int*)  d_in[7];
    const float* cover  = (const float*)d_in[8];
    // d_in[9] = step (unused)
    const float* emb    = (const float*)d_in[10];
    const float* Wx     = (const float*)d_in[11];
    const float* bx     = (const float*)d_in[12];
    const float* Wih    = (const float*)d_in[13];
    const float* Whh    = (const float*)d_in[14];
    const float* bih    = (const float*)d_in[15];
    const float* bhh    = (const float*)d_in[16];
    const float* W_h    = (const float*)d_in[17];
    const float* Wdp    = (const float*)d_in[18];
    const float* bdp    = (const float*)d_in[19];
    const float* W_c    = (const float*)d_in[20];
    const float* v_w    = (const float*)d_in[21];
    const float* Wpg    = (const float*)d_in[22];
    const float* bpg    = (const float*)d_in[23];
    const float* Wo1    = (const float*)d_in[24];
    const float* bo1    = (const float*)d_in[25];
    const float* Wo2    = (const float*)d_in[26];
    const float* bo2    = (const float*)d_in[27];

    float* o = (float*)d_out;
    float* fin  = o + O_FINAL;
    float* h_t  = o + O_HT;
    float* c_cc = o + O_CC;
    float* c_t  = o + O_CT;
    float* attn = o + O_ATTN;
    float* pgen = o + O_PGEN;
    float* covo = o + O_COV;

    float* w = (float*)d_ws;
    float* xbuf   = w + W_X;
    float* gates  = w + W_GATES;
    float* decf   = w + W_DECF;
    float* scores = w + W_SCORES;
    float* rmax   = w + W_RMAX;
    float* rsum   = w + W_RSUM;
    float* logits = w + W_LOGITS;
    _Float16* hbase = (_Float16*)(w + WS_FLOATS_END);
    _Float16* wh16  = hbase + Hx_WH16;
    _Float16* wo216 = hbase + Hx_WO216;
    _Float16* out16 = hbase + Hx_OUT16;

    // weight conversions (independent of everything else)
    k_cvt4   <<<(N * N / 4 + 255) / 256, 256, 0, stream>>>(W_h, wh16, N * N / 4);
    k_cvt4   <<<(V * H / 4 + 255) / 256, 256, 0, stream>>>(Wo2, wo216, V * H / 4);

    k_xproj  <<<(B * E + 255) / 256, 256, 0, stream>>>(y_t_1, c_t_1, emb, Wx, bx, xbuf);
    k_gates  <<<(B * G4H + 255) / 256, 256, 0, stream>>>(xbuf, h0, Wih, Whh, bih, bhh, gates);
    k_lstm   <<<(B * H + 255) / 256, 256, 0, stream>>>(gates, c0, h_t, c_cc);
    k_decfea <<<(B * N + 255) / 256, 256, 0, stream>>>(h_t, c_cc, Wdp, bdp, decf);
    // 6400 waves / 4 waves per block = 1600 blocks
    k_scores <<<1600, 128, 0, stream>>>(enc, wh16, decf, W_c, v_w, cover, scores);
    k_attn   <<<B, 256, 0, stream>>>(scores, mask, cover, attn, covo);
    k_ctx    <<<(B * N + 255) / 256, 256, 0, stream>>>(attn, enc, c_t);
    k_pgen   <<<B, 256, 0, stream>>>(c_t, h_t, c_cc, xbuf, Wpg, bpg, pgen);
    k_outo1  <<<(B * H + 255) / 256, 256, 0, stream>>>(h_t, c_t, Wo1, bo1, out16);
    k_logits <<<6250, 256, 0, stream>>>(out16, wo216, bo2, logits);
    k_vstats <<<B, 256, 0, stream>>>(logits, rmax, rsum);
    k_final  <<<(B * VO + 255) / 256, 256, 0, stream>>>(logits, rmax, rsum, pgen, xz, fin);
    k_scatter<<<(B * TK + 255) / 256, 256, 0, stream>>>(ev, attn, pgen, fin);
}
